// MoiraiMoEModule_85925115724419
// MI455X (gfx1250) — matrix operand order repaired and checked by reading, not compiled, not run on hardware
//
#include <hip/hip_runtime.h>
#include <hip/hip_bf16.h>

typedef _Float16 f16;
typedef __attribute__((ext_vector_type(16))) _Float16 v16h;
typedef __attribute__((ext_vector_type(8)))  float    v8f;
typedef __attribute__((__vector_size__(16))) int      i32x4;

#define TT   2048   // B*S
#define SS   512
#define BB   4
#define DD   384
#define HH   6
#define HDD  64
#define LL   4
#define DFFC 1536
#define EE   8

#if __has_builtin(__builtin_amdgcn_global_load_async_to_lds_b128)
#define HAS_ASYNC_LDS 1
#define AS3 __attribute__((address_space(3)))
#else
#define HAS_ASYNC_LDS 0
#endif

__device__ __forceinline__ void wait_async0() {
#if __has_builtin(__builtin_amdgcn_s_wait_asynccnt)
  __builtin_amdgcn_s_wait_asynccnt(0);
#else
  asm volatile("s_wait_asynccnt 0" ::: "memory");
#endif
}

// ---------------------------------------------------------------------------
// WMMA helpers (CDNA5 wave32, 16x16x32 f16 -> f32)
// ---------------------------------------------------------------------------
__device__ __forceinline__ v8f wmma32(v16h a, v16h b, v8f c) {
  return __builtin_amdgcn_wmma_f32_16x16x32_f16(false, a, false, b, (short)0, c,
                                                false, false);
}

// A fragment 16x32 (MxK), element (m,k) at base[m*ld + k].
// Lane L (lm=L&15, hi=L>>4): M=lm; holds k = 16*hi + e.
__device__ __forceinline__ v16h ldA(const f16* base, int ld) {
  int lane = threadIdx.x & 31;
  int lm = lane & 15, hi = lane >> 4;
  const f16* row = base + lm * ld + 16 * hi;
  v16h a;
#pragma unroll
  for (int e = 0; e < 16; ++e) a[e] = row[e];
  return a;
}

// B fragment 32x16 (KxN), element (k,n) at base[k*sk + n*sn].
// Lane L: N=lm; holds k = 16*hi + e.
__device__ __forceinline__ v16h ldB_f16(const f16* base, int sk, int sn) {
  int lane = threadIdx.x & 31;
  int lm = lane & 15, hi = lane >> 4;
  const f16* col = base + lm * sn + (16 * hi) * sk;
  v16h b;
#pragma unroll
  for (int e = 0; e < 16; ++e) b[e] = col[e * sk];
  return b;
}

// Same but streams f32 weights and converts to f16 in-register.
__device__ __forceinline__ v16h ldB_f32(const float* base, int sk, int sn) {
  int lane = threadIdx.x & 31;
  int lm = lane & 15, hi = lane >> 4;
  const float* col = base + lm * sn + (16 * hi) * sk;
  v16h b;
#pragma unroll
  for (int e = 0; e < 16; ++e) b[e] = (f16)col[e * sk];
  return b;
}

// ---------------------------------------------------------------------------
// Stats: per-(sample,variate) segment mean/var via atomics
// ---------------------------------------------------------------------------
__global__ void k_stats(const float* __restrict__ target, const unsigned char* __restrict__ obs,
                        const int* __restrict__ sid, const int* __restrict__ vid,
                        const unsigned char* __restrict__ pm, const int* __restrict__ pidx,
                        float* segCnt, float* segSum, float* segSq) {
  int t = blockIdx.x * blockDim.x + threadIdx.x;
  if (t >= TT) return;
  int p = pidx[t];
  int psz = (p == 0) ? 16 : ((p == 1) ? 32 : 64);
  bool pred = pm[t] != 0;
  float c = 0.f, s = 0.f, q = 0.f;
  for (int j = 0; j < 64; ++j) {
    float mm = (obs[t * 64 + j] && (j < psz) && !pred) ? 1.f : 0.f;
    float v = target[t * 64 + j];
    c += mm; s += v * mm; q += v * v * mm;
  }
  int b = t >> 9;
  int g = (b * 4 + sid[t]) * 4 + vid[t];
  atomicAdd(&segCnt[g], c);
  atomicAdd(&segSum[g], s);
  atomicAdd(&segSq[g], q);
}

__global__ void k_stats_fin(const float* segCnt, const float* segSum, const float* segSq,
                            float* locG, float* scaleG) {
  int g = threadIdx.x;
  if (g >= 64) return;
  float tobs = fmaxf(segCnt[g], 1.f);
  float loc = segSum[g] / tobs;
  float var = fmaxf(segSq[g] / tobs - loc * loc, 0.f);
  locG[g] = loc;
  scaleG[g] = sqrtf(var + 1e-5f);
}

// ---------------------------------------------------------------------------
// Embedding: h = W_feat[p]@silu(W_in[p]@x+b) + b_feat + W_res[p]@x + b_res
// ---------------------------------------------------------------------------
__global__ void k_embed(const float* __restrict__ target, const int* __restrict__ sid,
                        const int* __restrict__ vid, const int* __restrict__ pidx,
                        const float* __restrict__ locG, const float* __restrict__ scaleG,
                        const float* __restrict__ w_in, const float* __restrict__ b_in,
                        const float* __restrict__ w_feat, const float* __restrict__ b_feat,
                        const float* __restrict__ w_res, const float* __restrict__ b_res,
                        float* __restrict__ h, float* locT, float* scaleT) {
  int t = blockIdx.x;
  int b = t >> 9;
  int p = pidx[t];
  int psz = (p == 0) ? 16 : ((p == 1) ? 32 : 64);
  int g = (b * 4 + sid[t]) * 4 + vid[t];
  float lc = locG[g], sc = scaleG[g];
  __shared__ float Xs[64];
  __shared__ float IR[DD];
  int tid = threadIdx.x;
  if (tid == 0) { locT[t] = lc; scaleT[t] = sc; }
  if (tid < 64) {
    float v = target[t * 64 + tid];
    Xs[tid] = (tid < psz) ? (v - lc) / sc : 0.f;
  }
  __syncthreads();
  for (int o = tid; o < DD; o += 128) {
    const float* wr = w_in + (p * DD + o) * 64;
    float acc = b_in[p * DD + o];
    for (int i = 0; i < 64; ++i) acc += Xs[i] * wr[i];
    IR[o] = acc / (1.f + __expf(-acc));
  }
  __syncthreads();
  for (int o = tid; o < DD; o += 128) {
    const float* wf = w_feat + (p * DD + o) * DD;
    float acc = b_feat[p * DD + o];
    for (int d = 0; d < DD; ++d) acc += IR[d] * wf[d];
    const float* wr2 = w_res + (p * DD + o) * 64;
    float acc2 = b_res[p * DD + o];
    for (int i = 0; i < 64; ++i) acc2 += Xs[i] * wr2[i];
    h[t * DD + o] = acc + acc2;
  }
}

// ---------------------------------------------------------------------------
// RMSNorm (f32 in -> f16 out)
// ---------------------------------------------------------------------------
__global__ void k_rmsnorm_f16(const float* __restrict__ X, const float* __restrict__ Wt,
                              f16* __restrict__ Y) {
  int t = blockIdx.x;
  __shared__ float red[128];
  float ss = 0.f;
  for (int d = threadIdx.x; d < DD; d += 128) { float v = X[t * DD + d]; ss += v * v; }
  red[threadIdx.x] = ss;
  __syncthreads();
  for (int st = 64; st > 0; st >>= 1) {
    if (threadIdx.x < st) red[threadIdx.x] += red[threadIdx.x + st];
    __syncthreads();
  }
  float rs = rsqrtf(red[0] / (float)DD + 1e-5f);
  for (int d = threadIdx.x; d < DD; d += 128) Y[t * DD + d] = (f16)(X[t * DD + d] * rs * Wt[d]);
}

// ---------------------------------------------------------------------------
// Generic WMMA GEMM: Y[MxN](f32) (+)= A[MxK](f16,row-major) * W[NxK](f32)^T
// One 16x16 output tile per wave, 8 waves / block.
// ---------------------------------------------------------------------------
__global__ void __launch_bounds__(256) k_gemm(const f16* __restrict__ A,
                                              const float* __restrict__ W,
                                              float* __restrict__ Y,
                                              int M, int N, int K, int accum) {
  int wave = threadIdx.x >> 5;
  int tiles_n = N >> 4;
  int tile = blockIdx.x * 8 + wave;
  if (tile >= (M >> 4) * tiles_n) return;
  int tm = tile / tiles_n, tn = tile % tiles_n;
  const f16* Ab = A + (tm * 16) * K;
  const float* Wb = W + (tn * 16) * K;
  v8f c = {};
  for (int kb = 0; kb < K; kb += 32) {
    // pull the next weight k-block toward GL2 while this one feeds the WMMA pipe
    if (kb + 32 < K) __builtin_prefetch(Wb + ((threadIdx.x & 15) * K) + kb + 32, 0, 1);
    c = wmma32(ldA(Ab + kb, K), ldB_f32(Wb + kb, 1, K), c);
  }
  int lane = threadIdx.x & 31;
  int lm = lane & 15, hi = lane >> 4;
#pragma unroll
  for (int r = 0; r < 8; ++r) {
    int m = tm * 16 + r + 8 * hi, n = tn * 16 + lm;
    if (accum) Y[m * N + n] += c[r]; else Y[m * N + n] = c[r];
  }
}

// ---------------------------------------------------------------------------
// Per-head QK rmsnorm + RoPE + V f16 conversion. One (t,h) per 64-thread block.
// ---------------------------------------------------------------------------
__global__ void k_qkprep(const float* __restrict__ qf, const float* __restrict__ kf,
                         const float* __restrict__ vf, const float* __restrict__ qnw,
                         const float* __restrict__ knw, const int* __restrict__ time_id,
                         f16* __restrict__ q16, f16* __restrict__ k16, f16* __restrict__ v16o) {
  int th = blockIdx.x;
  int t = th / HH, hh = th % HH;
  int d = threadIdx.x;
  int base = t * DD + hh * HDD;
  __shared__ float buf[64], red[64];
  float tm = (float)time_id[t];
  for (int which = 0; which < 2; ++which) {
    const float* src = which ? kf : qf;
    const float* wn = which ? knw : qnw;
    f16* dst = which ? k16 : q16;
    float v = src[base + d];
    red[d] = v * v;
    __syncthreads();
    for (int st = 32; st > 0; st >>= 1) {
      if (d < st) red[d] += red[d + st];
      __syncthreads();
    }
    float rs = rsqrtf(red[0] / (float)HDD + 1e-5f);
    float zn = v * rs * wn[d];
    buf[d] = zn;
    __syncthreads();
    float outv;
    if (d < 16) {
      float ang = tm * __powf(10000.f, -(float)d / 16.f);
      outv = zn * __cosf(ang) - buf[d + 16] * __sinf(ang);
    } else if (d < 32) {
      float ang = tm * __powf(10000.f, -(float)(d - 16) / 16.f);
      outv = zn * __cosf(ang) + buf[d - 16] * __sinf(ang);
    } else outv = zn;
    dst[base + d] = (f16)outv;
    __syncthreads();
  }
  v16o[base + d] = (f16)vf[base + d];
}

// ---------------------------------------------------------------------------
// Attention: one block per (q-tile16, head, batch). Full K/V head staged in LDS
// (320KB/WGP budget) via async-to-LDS copies. QK^T + bias/mask + softmax + PV,
// all WMMA f16.
// ---------------------------------------------------------------------------
#define ATT_LDS (16*SS*4 + (128+128+16+16)*4 + 2*SS*HDD*2 + 16*SS*2 + 3*SS*4)
__global__ void k_attn(const f16* __restrict__ q16, const f16* __restrict__ k16,
                       const f16* __restrict__ v16i, const int* __restrict__ sid,
                       const int* __restrict__ tidv, const int* __restrict__ vidv,
                       const float* __restrict__ var_bias_l, f16* __restrict__ o16) {
  int qt = blockIdx.x, hh = blockIdx.y, b = blockIdx.z;
  extern __shared__ char smem[];
  float* Ssc  = (float*)smem;             // 16 x 512
  float* pmax = Ssc + 16 * SS;            // 128
  float* psum = pmax + 128;               // 128
  float* rmax = psum + 128;               // 16
  float* rsum = rmax + 16;                // 16 (reciprocal sums)
  f16* Kh = (f16*)(rsum + 16);            // 512 x 64
  f16* Vh = Kh + SS * HDD;                // 512 x 64
  f16* Pp = Vh + SS * HDD;                // 16 x 512
  int* sS = (int*)(Pp + 16 * SS);
  int* tS = sS + SS;
  int* vS = tS + SS;

  int tid = threadIdx.x;  // 128 threads = 4 waves
#if HAS_ASYNC_LDS
  // 16B-per-lane async DMA of the whole K/V head into LDS (ASYNCcnt-tracked)
  for (int idx = tid * 8; idx < SS * HDD; idx += 128 * 8) {
    int s = idx >> 6, d = idx & 63;
    const f16* gk = k16 + (b * SS + s) * DD + hh * HDD + d;
    const f16* gv = v16i + (b * SS + s) * DD + hh * HDD + d;
    __builtin_amdgcn_global_load_async_to_lds_b128((i32x4*)gk, (AS3 i32x4*)(Kh + idx), 0, 0);
    __builtin_amdgcn_global_load_async_to_lds_b128((i32x4*)gv, (AS3 i32x4*)(Vh + idx), 0, 0);
  }
#else
  for (int idx = tid; idx < SS * HDD; idx += 128) {
    int s = idx >> 6, d = idx & 63;
    int g = (b * SS + s) * DD + hh * HDD + d;
    Kh[idx] = k16[g];
    Vh[idx] = v16i[g];
  }
#endif
  for (int s = tid; s < SS; s += 128) {
    sS[s] = sid[b * SS + s]; tS[s] = tidv[b * SS + s]; vS[s] = vidv[b * SS + s];
  }
#if HAS_ASYNC_LDS
  wait_async0();
#endif
  __syncthreads();

  int wave = tid >> 5, lane = tid & 31;
  int lm = lane & 15, hi = lane >> 4;
  float vb0 = var_bias_l[hh * 2 + 0], vb1 = var_bias_l[hh * 2 + 1];

  const f16* Qb = q16 + (b * SS + qt * 16) * DD + hh * HDD;
  v16h a0 = ldA(Qb, DD);
  v16h a1 = ldA(Qb + 32, DD);
  for (int kt = wave; kt < SS / 16; kt += 4) {
    v8f c = {};
    c = wmma32(a0, ldB_f16(Kh + kt * 16 * HDD, 1, HDD), c);
    c = wmma32(a1, ldB_f16(Kh + kt * 16 * HDD + 32, 1, HDD), c);
    int n = kt * 16 + lm;
#pragma unroll
    for (int r = 0; r < 8; ++r) {
      int m = r + 8 * hi;
      int qg = qt * 16 + m;
      bool msk = (sS[qg] == sS[n]) && (tS[qg] >= tS[n]);
      float bias = (vS[qg] == vS[n]) ? vb0 : vb1;
      Ssc[m * SS + n] = msk ? (c[r] * 0.125f + bias) : -1e9f;
    }
  }
  __syncthreads();

  // parallel row softmax over 512 keys
  int row = tid >> 3, seg = tid & 7;
  float mx = -1e30f;
  for (int n2 = seg; n2 < SS; n2 += 8) mx = fmaxf(mx, Ssc[row * SS + n2]);
  pmax[tid] = mx;
  __syncthreads();
  if (tid < 16) {
    float m2 = pmax[tid * 8];
    for (int g2 = 1; g2 < 8; ++g2) m2 = fmaxf(m2, pmax[tid * 8 + g2]);
    rmax[tid] = m2;
  }
  __syncthreads();
  float rm = rmax[row], sm = 0.f;
  for (int n2 = seg; n2 < SS; n2 += 8) sm += __expf(Ssc[row * SS + n2] - rm);
  psum[tid] = sm;
  __syncthreads();
  if (tid < 16) {
    float s2 = 0.f;
    for (int g2 = 0; g2 < 8; ++g2) s2 += psum[tid * 8 + g2];
    rsum[tid] = 1.f / s2;
  }
  __syncthreads();
  for (int idx = tid; idx < 16 * SS; idx += 128) {
    int m = idx >> 9;
    Pp[idx] = (f16)(__expf(Ssc[idx] - rmax[m]) * rsum[m]);
  }
  __syncthreads();

  // P(16x512) @ V(512x64): wave -> one 16-col output tile
  v8f acc = {};
  for (int ks = 0; ks < SS / 32; ++ks)
    acc = wmma32(ldA(Pp + ks * 32, SS), ldB_f16(Vh + ks * 32 * HDD + wave * 16, HDD, 1), acc);
#pragma unroll
  for (int r = 0; r < 8; ++r) {
    int m = r + 8 * hi;
    o16[(b * SS + qt * 16 + m) * DD + hh * HDD + wave * 16 + lm] = (f16)acc[r];
  }
}

// ---------------------------------------------------------------------------
// Router: softmax(z @ Wr^T), top-2, normalized gates, build expert lists.
// One wave per token.
// ---------------------------------------------------------------------------
__global__ void k_router(const f16* __restrict__ z16, const float* __restrict__ rw,
                         int* cnt, int* elist, float* egate) {
  int wave = threadIdx.x >> 5, lane = threadIdx.x & 31;
  int t = blockIdx.x * 8 + wave;
  if (t >= TT) return;
  float pl[EE];
#pragma unroll
  for (int e = 0; e < EE; ++e) {
    float s = 0.f;
    for (int d = lane; d < DD; d += 32) s += (float)z16[t * DD + d] * rw[e * DD + d];
    pl[e] = s;
  }
#pragma unroll
  for (int e = 0; e < EE; ++e)
    for (int off = 16; off > 0; off >>= 1) pl[e] += __shfl_xor(pl[e], off, 32);
  if (lane == 0) {
    float mx = pl[0];
    for (int e = 1; e < EE; ++e) mx = fmaxf(mx, pl[e]);
    float pr[EE], sum = 0.f;
    for (int e = 0; e < EE; ++e) { pr[e] = __expf(pl[e] - mx); sum += pr[e]; }
    for (int e = 0; e < EE; ++e) pr[e] /= sum;
    int i0 = 0;
    for (int e = 1; e < EE; ++e) if (pr[e] > pr[i0]) i0 = e;
    int i1 = (i0 == 0) ? 1 : 0;
    for (int e = 0; e < EE; ++e) if (e != i0 && pr[e] > pr[i1]) i1 = e;
    float gs = pr[i0] + pr[i1];
    int p0 = atomicAdd(&cnt[i0], 1);
    elist[i0 * TT + p0] = t; egate[i0 * TT + p0] = pr[i0] / gs;
    int p1 = atomicAdd(&cnt[i1], 1);
    elist[i1 * TT + p1] = t; egate[i1 * TT + p1] = pr[i1] / gs;
  }
}

// ---------------------------------------------------------------------------
// Fused MoE FFN per (expert, 16-token tile): a = silu(Z W1^T) * (Z W3^T) in LDS,
// then Y = a W2^T, gate-scaled atomic scatter into h. Weights stream f32->f16.
// ---------------------------------------------------------------------------
__global__ void __launch_bounds__(256) k_moe(const f16* __restrict__ z16, const int* __restrict__ cnt,
                                             const int* __restrict__ elist, const float* __restrict__ egate,
                                             const float* __restrict__ w1e_, const float* __restrict__ w2e_,
                                             const float* __restrict__ w3e_, float* __restrict__ h) {
  int e = blockIdx.y, tile = blockIdx.x;
  int count = cnt[e];
  int base = tile * 16;
  if (base >= count) return;
  __shared__ f16 Zt[16 * DD];
  __shared__ f16 At[16 * DFFC];
  __shared__ int toks[16];
  __shared__ float gts[16];
  int tid = threadIdx.x;
  if (tid < 16) {
    int i = base + tid;
    toks[tid] = (i < count) ? elist[e * TT + i] : 0;
    gts[tid]  = (i < count) ? egate[e * TT + i] : 0.f;
  }
  __syncthreads();
  for (int idx = tid; idx < 16 * DD; idx += 256) {
    int m = idx / DD, d = idx - m * DD;
    Zt[idx] = z16[toks[m] * DD + d];
  }
  __syncthreads();
  int wave = tid >> 5, lane = tid & 31;
  int lm = lane & 15, hi = lane >> 4;
  const float* w1p = w1e_ + (size_t)e * DFFC * DD;
  const float* w3p = w3e_ + (size_t)e * DFFC * DD;
  const float* w2p = w2e_ + (size_t)e * DD * DFFC;
  for (int nt = wave; nt < DFFC / 16; nt += 8) {
    v8f c1 = {}, c3 = {};
    for (int kb = 0; kb < DD; kb += 32) {
      if (kb + 32 < DD) {
        __builtin_prefetch(w1p + (nt * 16 + (threadIdx.x & 15)) * DD + kb + 32, 0, 1);
        __builtin_prefetch(w3p + (nt * 16 + (threadIdx.x & 15)) * DD + kb + 32, 0, 1);
      }
      v16h a = ldA(Zt + kb, DD);
      c1 = wmma32(a, ldB_f32(w1p + nt * 16 * DD + kb, 1, DD), c1);
      c3 = wmma32(a, ldB_f32(w3p + nt * 16 * DD + kb, 1, DD), c3);
    }
#pragma unroll
    for (int r = 0; r < 8; ++r) {
      float u = c1[r];
      float aa = (u / (1.f + __expf(-u))) * c3[r];
      At[(r + 8 * hi) * DFFC + nt * 16 + lm] = (f16)aa;
    }
  }
  __syncthreads();
  for (int nt = wave; nt < DD / 16; nt += 8) {
    v8f acc = {};
    for (int kb = 0; kb < DFFC; kb += 32) {
      if (kb + 32 < DFFC)
        __builtin_prefetch(w2p + (nt * 16 + (threadIdx.x & 15)) * DFFC + kb + 32, 0, 1);
      acc = wmma32(ldA(At + kb, DFFC), ldB_f32(w2p + nt * 16 * DFFC + kb, 1, DFFC), acc);
    }
#pragma unroll
    for (int r = 0; r < 8; ++r) {
      int m = r + 8 * hi;
      atomicAdd(&h[toks[m] * DD + nt * 16 + lm], gts[m] * acc[r]);
    }
  }
}

// ---------------------------------------------------------------------------
// Output head: pp = W_out[p] @ h_norm + b_out[p]; mu/sigma denormalize.
// ---------------------------------------------------------------------------
__global__ void k_head(const f16* __restrict__ zn, const int* __restrict__ pidx,
                       const float* __restrict__ w_out, const float* __restrict__ b_out,
                       const float* __restrict__ locT, const float* __restrict__ scaleT,
                       float* __restrict__ out) {
  int t = blockIdx.x;
  int p = pidx[t];
  __shared__ float Z[DD];
  int tid = threadIdx.x;
  for (int d = tid; d < DD; d += 128) Z[d] = (float)zn[t * DD + d];
  __syncthreads();
  int o = tid;  // 0..127
  const float* wr = w_out + (p * 128 + o) * DD;
  float acc = b_out[p * 128 + o];
  for (int d = 0; d < DD; ++d) acc += Z[d] * wr[d];
  float sc = scaleT[t], lc = locT[t];
  if (o < 64) {
    out[t * 64 + o] = acc * sc + lc;
  } else {
    float sp = (acc > 20.f) ? acc : log1pf(__expf(acc));
    out[TT * 64 + t * 64 + (o - 64)] = sp * sc;
  }
}

// ---------------------------------------------------------------------------
extern "C" void kernel_launch(void* const* d_in, const int* in_sizes, int n_in,
                              void* d_out, int out_size, void* d_ws, size_t ws_size,
                              hipStream_t stream) {
  const float* target          = (const float*)d_in[0];
  const unsigned char* obs     = (const unsigned char*)d_in[1];
  const int* sample_id         = (const int*)d_in[2];
  const int* time_id           = (const int*)d_in[3];
  const int* variate_id        = (const int*)d_in[4];
  const unsigned char* predm   = (const unsigned char*)d_in[5];
  const int* patch_idx         = (const int*)d_in[6];
  const float* w_in            = (const float*)d_in[7];
  const float* b_in            = (const float*)d_in[8];
  const float* w_feat          = (const float*)d_in[9];
  const float* b_feat          = (const float*)d_in[10];
  const float* w_res            = (const float*)d_in[11];
  const float* b_res           = (const float*)d_in[12];
  const float* norm1_w         = (const float*)d_in[13];
  const float* norm2_w         = (const float*)d_in[14];
  const float* wq              = (const float*)d_in[15];
  const float* wk              = (const float*)d_in[16];
  const float* wv              = (const float*)d_in[17];
  const float* wo              = (const float*)d_in[18];
  const float* qnorm_w         = (const float*)d_in[19];
  const float* knorm_w         = (const float*)d_in[20];
  const float* var_bias        = (const float*)d_in[21];
  const float* router_w        = (const float*)d_in[22];
  const float* w1              = (const float*)d_in[23];
  const float* w2              = (const float*)d_in[24];
  const float* w3              = (const float*)d_in[25];
  const float* final_norm_w    = (const float*)d_in[26];
  const float* w_out           = (const float*)d_in[27];
  const float* b_out           = (const float*)d_in[28];

  char* wp = (char*)d_ws;
  auto alloc = [&](size_t bytes) -> char* {
    char* r = wp;
    wp += (bytes + 255) & ~(size_t)255;
    return r;
  };
  float* segCnt = (float*)alloc(3 * 64 * sizeof(float));
  float* segSum = segCnt + 64;
  float* segSq  = segCnt + 128;
  float* locG   = (float*)alloc(64 * sizeof(float));
  float* scaleG = (float*)alloc(64 * sizeof(float));
  float* locT   = (float*)alloc(TT * sizeof(float));
  float* scaleT = (float*)alloc(TT * sizeof(float));
  float* h      = (float*)alloc((size_t)TT * DD * sizeof(float));
  f16*   hn     = (f16*)  alloc((size_t)TT * DD * sizeof(f16));
  float* qf     = (float*)alloc((size_t)TT * DD * sizeof(float));
  float* kf     = (float*)alloc((size_t)TT * DD * sizeof(float));
  float* vf     = (float*)alloc((size_t)TT * DD * sizeof(float));
  f16*   q16    = (f16*)  alloc((size_t)TT * DD * sizeof(f16));
  f16*   k16    = (f16*)  alloc((size_t)TT * DD * sizeof(f16));
  f16*   v16b   = (f16*)  alloc((size_t)TT * DD * sizeof(f16));
  f16*   o16    = (f16*)  alloc((size_t)TT * DD * sizeof(f16));
  f16*   z16    = (f16*)  alloc((size_t)TT * DD * sizeof(f16));
  int*   ecnt   = (int*)  alloc(EE * sizeof(int));
  int*   elist  = (int*)  alloc((size_t)EE * TT * sizeof(int));
  float* egate  = (float*)alloc((size_t)EE * TT * sizeof(float));

  (void)hipMemsetAsync(segCnt, 0, 3 * 64 * sizeof(float), stream);
  k_stats<<<TT / 256, 256, 0, stream>>>(target, obs, sample_id, variate_id, predm,
                                        patch_idx, segCnt, segSum, segSq);
  k_stats_fin<<<1, 64, 0, stream>>>(segCnt, segSum, segSq, locG, scaleG);
  k_embed<<<TT, 128, 0, stream>>>(target, sample_id, variate_id, patch_idx, locG, scaleG,
                                  w_in, b_in, w_feat, b_feat, w_res, b_res, h, locT, scaleT);

  int gemm_blocks = ((TT / 16) * (DD / 16)) / 8;  // 384
  for (int l = 0; l < LL; ++l) {
    k_rmsnorm_f16<<<TT, 128, 0, stream>>>(h, norm1_w + l * DD, hn);
    k_gemm<<<gemm_blocks, 256, 0, stream>>>(hn, wq + l * DD * DD, qf, TT, DD, DD, 0);
    k_gemm<<<gemm_blocks, 256, 0, stream>>>(hn, wk + l * DD * DD, kf, TT, DD, DD, 0);
    k_gemm<<<gemm_blocks, 256, 0, stream>>>(hn, wv + l * DD * DD, vf, TT, DD, DD, 0);
    k_qkprep<<<TT * HH, 64, 0, stream>>>(qf, kf, vf, qnorm_w + l * HDD, knorm_w + l * HDD,
                                         time_id, q16, k16, v16b);
    dim3 agrid(SS / 16, HH, BB);
    k_attn<<<agrid, 128, ATT_LDS, stream>>>(q16, k16, v16b, sample_id, time_id, variate_id,
                                            var_bias + l * HH * 2, o16);
    k_gemm<<<gemm_blocks, 256, 0, stream>>>(o16, wo + l * DD * DD, h, TT, DD, DD, 1);
    k_rmsnorm_f16<<<TT, 128, 0, stream>>>(h, norm2_w + l * DD, z16);
    (void)hipMemsetAsync(ecnt, 0, EE * sizeof(int), stream);
    k_router<<<TT / 8, 256, 0, stream>>>(z16, router_w + l * EE * DD, ecnt, elist, egate);
    dim3 mgrid(TT / 16, EE);
    k_moe<<<mgrid, 256, 0, stream>>>(z16, ecnt, elist, egate,
                                     w1 + (size_t)l * EE * DFFC * DD,
                                     w2 + (size_t)l * EE * DD * DFFC,
                                     w3 + (size_t)l * EE * DFFC * DD, h);
  }
  k_rmsnorm_f16<<<TT, 128, 0, stream>>>(h, final_norm_w, z16);
  k_head<<<TT, 128, 0, stream>>>(z16, patch_idx, w_out, b_out, locT, scaleT, (float*)d_out);
}
